// GNNStage_43774306681045
// MI455X (gfx1250) — matrix operand order repaired and checked by reading, not compiled, not run on hardware
//
#include <hip/hip_runtime.h>
#include <hip/hip_bf16.h>
#include <math.h>

typedef __attribute__((ext_vector_type(16))) __bf16 v16bf;
typedef __attribute__((ext_vector_type(8)))  float  v8f;
typedef int v4i_t __attribute__((vector_size(16)));   // matches builtin param type

#define NNODES 30000

#if defined(__gfx1250__) && __has_builtin(__builtin_amdgcn_global_load_async_to_lds_b128) && __has_builtin(__builtin_amdgcn_s_wait_asynccnt)
#define HAVE_ASYNC_LDS 1
#else
#define HAVE_ASYNC_LDS 0
#endif

// ---------------- helpers ----------------
__device__ __forceinline__ unsigned bfbits(float f) {   // bf16 bits, round-to-nearest-even
  unsigned u = __builtin_bit_cast(unsigned, f);
  return ((u + 0x7FFFu + ((u >> 16) & 1u)) >> 16) & 0xFFFFu;
}
__device__ __forceinline__ float bits2f(unsigned hi16) {
  unsigned u = hi16 << 16;
  return __builtin_bit_cast(float, u);
}
__device__ __forceinline__ float wred32(float x) {
  #pragma unroll
  for (int o = 16; o > 0; o >>= 1) x += __shfl_xor(x, o, 32);
  return x;
}
__device__ __forceinline__ float gelu_exact(float x) {
  return 0.5f * x * (1.0f + erff(x * 0.70710678118654752440f));
}

struct U8x32 { uint4 a, b; };
__device__ __forceinline__ v16bf frag_cast(uint4 a, uint4 b) {
  U8x32 t; t.a = a; t.b = b;
  return __builtin_bit_cast(v16bf, t);
}

// ---------------- pre-pass: fp32 -> fragment-swizzled packed bf16 hi/lo ----------------
// Tile = 16 rows x 32 k. Per tile: 512 dwords laid out as [g 0..3][lane 0..31][dw 0..3]:
//   g0,g1 = hi pairs (p=0..7), g2,g3 = lo pairs. dword p packs elements e=2p,2p+1.
// mode 0 = A-operand k-mapping, mode 1 = B-operand k-mapping (ISA 7.12.2 layouts).
__global__ __launch_bounds__(256) void swizzle_pack(
    const float* __restrict__ src, unsigned* __restrict__ dst,
    int Mrows, int K, int nkt, int mode)
{
  long long i = (long long)blockIdx.x * 256 + threadIdx.x;
  int Mt = (Mrows + 15) >> 4;
  long long total = (long long)Mt * nkt * 256;
  if (i >= total) return;
  int tile = (int)(i >> 8);
  int rem  = (int)(i & 255);
  int lane = rem >> 3;
  int p    = rem & 7;
  int mt = tile / nkt, kt = tile - mt * nkt;
  int row = mt * 16 + (lane & 15);
  int e = 2 * p;
  int kk;
  if (mode == 0) kk = ((e & 8) << 1) + ((lane >> 4) << 3) + (e & 7);
  else           kk = ((e & 8) << 1) + ((lane >> 4) << 3) + (e & 7);
  int k = kt * 32 + kk;
  float f0 = (row < Mrows && k     < K) ? src[(size_t)row * K + k]     : 0.f;
  float f1 = (row < Mrows && k + 1 < K) ? src[(size_t)row * K + k + 1] : 0.f;
  unsigned h0 = bfbits(f0), h1 = bfbits(f1);
  unsigned l0 = bfbits(f0 - bits2f(h0)), l1 = bfbits(f1 - bits2f(h1));
  size_t base = (size_t)tile * 512;
  int g = p >> 2, d = p & 3;
  dst[base + ((size_t)(g)     * 32 + lane) * 4 + d] = h0 | (h1 << 16);
  dst[base + ((size_t)(g + 2) * 32 + lane) * 4 + d] = l0 | (l1 << 16);
}

// ---------------- WMMA GEMM on swizzled operands ----------------
// C[M,Nc] = A[M,K] @ W[Nc,K]^T + bias + addend, fp32 accum, bf16 hi/lo (3 wmma per frag pair).
// Block: 256 threads = 8 waves; 32 rows x 128 cols per block. Grid: (ceil(M/32), ceil(Nc/128)).
// A tile staged in LDS (async double-buffered when available); B fragments direct from global.
__global__ __launch_bounds__(256) void gemm_wmma3(
    const uint4* __restrict__ Asw, const uint4* __restrict__ Bsw,
    const float* __restrict__ bias, const float* __restrict__ addend,
    float* __restrict__ C, int M, int Nc, int nkt)
{
#if HAVE_ASYNC_LDS
  __shared__ uint4 sA[2][2][4][32];  // [buf][subtile][group][lane], 8 KB
#else
  __shared__ uint4 sA[1][2][4][32];  // 4 KB
#endif
  const int tid  = threadIdx.x;
  const int lane = tid & 31;
  const int wave = tid >> 5;
  const int mt0  = blockIdx.x * 2;            // first 16-row tile
  const int nt   = blockIdx.y * 8 + wave;     // 16-col tile index
  const int Mt   = (M + 15) >> 4;
  const int ntcnt = (Nc + 15) >> 4;
  const bool nvalid = (nt < ntcnt);

  // staging decomposition: one uint4 per thread (wave-uniform s_sub)
  const int s_sub = tid >> 7;
  const int s_g   = (tid >> 5) & 3;
  const int s_l   = tid & 31;
  const int mt_st = mt0 + s_sub;
  const bool stage_ok = (mt_st < Mt);

  v8f acc0 = {0.f,0.f,0.f,0.f,0.f,0.f,0.f,0.f};
  v8f acc1 = {0.f,0.f,0.f,0.f,0.f,0.f,0.f,0.f};

#if HAVE_ASYNC_LDS
  // tail subtile never written by async copies -> pre-zero its slots in both buffers
  if (!stage_ok) {
    uint4 z = make_uint4(0u, 0u, 0u, 0u);
    sA[0][s_sub][s_g][s_l] = z;
    sA[1][s_sub][s_g][s_l] = z;
  }
  auto issue = [&](int buf, int kt) {
    if (stage_ok) {
      const uint4* src = &Asw[((size_t)(mt_st * nkt + kt)) * 128 + s_g * 32 + s_l];
      __builtin_amdgcn_global_load_async_to_lds_b128(
          (__attribute__((address_space(1))) v4i_t*)src,
          (__attribute__((address_space(3))) v4i_t*)&sA[buf][s_sub][s_g][s_l],
          0, 0);
    }
  };
  issue(0, 0);
  if (nkt > 1) issue(1, 1);

  for (int kt = 0; kt < nkt; ++kt) {
    if (kt + 1 < nkt) __builtin_amdgcn_s_wait_asynccnt(1);
    else              __builtin_amdgcn_s_wait_asynccnt(0);
    __syncthreads();                       // tile kt visible to all waves
    const int buf = kt & 1;

    v16bf bhi, blo;
    if (nvalid) {
      size_t bt = ((size_t)(nt * nkt + kt)) * 128;
      bhi = frag_cast(Bsw[bt + lane],      Bsw[bt + 32 + lane]);
      blo = frag_cast(Bsw[bt + 64 + lane], Bsw[bt + 96 + lane]);
    }
    v16bf ahi0 = frag_cast(sA[buf][0][0][lane], sA[buf][0][1][lane]);
    v16bf alo0 = frag_cast(sA[buf][0][2][lane], sA[buf][0][3][lane]);
    v16bf ahi1 = frag_cast(sA[buf][1][0][lane], sA[buf][1][1][lane]);
    v16bf alo1 = frag_cast(sA[buf][1][2][lane], sA[buf][1][3][lane]);

    if (nvalid) {
      acc0 = __builtin_amdgcn_wmma_f32_16x16x32_bf16(false, ahi0, false, bhi, (short)0, acc0, false, false);
      acc0 = __builtin_amdgcn_wmma_f32_16x16x32_bf16(false, alo0, false, bhi, (short)0, acc0, false, false);
      acc0 = __builtin_amdgcn_wmma_f32_16x16x32_bf16(false, ahi0, false, blo, (short)0, acc0, false, false);
      acc1 = __builtin_amdgcn_wmma_f32_16x16x32_bf16(false, ahi1, false, bhi, (short)0, acc1, false, false);
      acc1 = __builtin_amdgcn_wmma_f32_16x16x32_bf16(false, alo1, false, bhi, (short)0, acc1, false, false);
      acc1 = __builtin_amdgcn_wmma_f32_16x16x32_bf16(false, ahi1, false, blo, (short)0, acc1, false, false);
    }
    __syncthreads();                       // all waves done with buf before reuse
    if (kt + 2 < nkt) issue(buf, kt + 2);  // prefetch into the buffer just freed
  }
#else
  for (int kt = 0; kt < nkt; ++kt) {
    uint4 val = make_uint4(0u, 0u, 0u, 0u);
    if (stage_ok)
      val = Asw[((size_t)(mt_st * nkt + kt)) * 128 + s_g * 32 + s_l];
    __syncthreads();
    sA[0][s_sub][s_g][s_l] = val;

    v16bf bhi, blo;
    if (nvalid) {
      size_t bt = ((size_t)(nt * nkt + kt)) * 128;
      bhi = frag_cast(Bsw[bt + lane],      Bsw[bt + 32 + lane]);
      blo = frag_cast(Bsw[bt + 64 + lane], Bsw[bt + 96 + lane]);
    }
    __syncthreads();

    v16bf ahi0 = frag_cast(sA[0][0][0][lane], sA[0][0][1][lane]);
    v16bf alo0 = frag_cast(sA[0][0][2][lane], sA[0][0][3][lane]);
    v16bf ahi1 = frag_cast(sA[0][1][0][lane], sA[0][1][1][lane]);
    v16bf alo1 = frag_cast(sA[0][1][2][lane], sA[0][1][3][lane]);

    if (nvalid) {
      acc0 = __builtin_amdgcn_wmma_f32_16x16x32_bf16(false, ahi0, false, bhi, (short)0, acc0, false, false);
      acc0 = __builtin_amdgcn_wmma_f32_16x16x32_bf16(false, alo0, false, bhi, (short)0, acc0, false, false);
      acc0 = __builtin_amdgcn_wmma_f32_16x16x32_bf16(false, ahi0, false, blo, (short)0, acc0, false, false);
      acc1 = __builtin_amdgcn_wmma_f32_16x16x32_bf16(false, ahi1, false, bhi, (short)0, acc1, false, false);
      acc1 = __builtin_amdgcn_wmma_f32_16x16x32_bf16(false, alo1, false, bhi, (short)0, acc1, false, false);
      acc1 = __builtin_amdgcn_wmma_f32_16x16x32_bf16(false, ahi1, false, blo, (short)0, acc1, false, false);
    }
  }
#endif

  // epilogue: D layout -> lane holds col n = lane&15, rows m = r + 8*(lane>=16)
  const int nc = nt * 16 + (lane & 15);
  if (!nvalid || nc >= Nc) return;
  float bval = bias ? bias[nc] : 0.f;
  const int mb = mt0 * 16 + ((lane >> 4) << 3);
  #pragma unroll
  for (int r = 0; r < 8; ++r) {
    int gm = mb + r;
    if (gm < M) {
      float v = acc0[r] + bval;
      if (addend) v += addend[(size_t)gm * Nc + nc];
      C[(size_t)gm * Nc + nc] = v;
    }
    int gm1 = gm + 16;
    if (gm1 < M) {
      float v = acc1[r] + bval;
      if (addend) v += addend[(size_t)gm1 * Nc + nc];
      C[(size_t)gm1 * Nc + nc] = v;
    }
  }
}

// ---------------- rowwise op: optional L2-normalize -> LayerNorm -> GELU, + addrow ----------------
// flags: 1=L2NORM, 2=LN, 4=GELU
__global__ __launch_bounds__(256) void row_op(
    const float* __restrict__ in, const float* __restrict__ g, const float* __restrict__ b,
    const float* __restrict__ addrow, float* __restrict__ out,
    int M, int Wd, int out_stride, int flags)
{
  int row  = blockIdx.x * 8 + (threadIdx.x >> 5);
  int lane = threadIdx.x & 31;
  if (row >= M) return;
  const float* x = in + (size_t)row * Wd;
  const int nper = Wd >> 5;  // Wd in {64,128,256}
  float v[8];
  #pragma unroll 8
  for (int i = 0; i < nper; ++i) v[i] = x[lane + (i << 5)];

  if (flags & 1) {
    float s = 0.f;
    for (int i = 0; i < nper; ++i) s += v[i] * v[i];
    s = wred32(s);
    float inv = 1.0f / fmaxf(sqrtf(s), 1e-12f);
    for (int i = 0; i < nper; ++i) v[i] *= inv;
  }
  if (flags & 2) {
    float s = 0.f;
    for (int i = 0; i < nper; ++i) s += v[i];
    float mu = wred32(s) / (float)Wd;
    float vs = 0.f;
    for (int i = 0; i < nper; ++i) { float d = v[i] - mu; vs += d * d; }
    float var = wred32(vs) / (float)Wd;
    float rs = rsqrtf(var + 1e-5f);
    for (int i = 0; i < nper; ++i) {
      int c = lane + (i << 5);
      v[i] = (v[i] - mu) * rs * g[c] + b[c];
    }
  }
  if (flags & 4) {
    for (int i = 0; i < nper; ++i) v[i] = gelu_exact(v[i]);
  }
  if (addrow) {
    for (int i = 0; i < nper; ++i) v[i] += addrow[lane + (i << 5)];
  }
  float* o = out + (size_t)row * out_stride;
  for (int i = 0; i < nper; ++i) o[lane + (i << 5)] = v[i];
}

// ---------------- 3-token multi-head attention (nh=4, dh=32), one wave per node ----------------
__global__ __launch_bounds__(256) void attn3_kernel(
    const float* __restrict__ qkv, float* __restrict__ hout, int N)
{
  int node = blockIdx.x * 8 + (threadIdx.x >> 5);
  int lane = threadIdx.x & 31;
  if (node >= N) return;
  const float* base = qkv + (size_t)node * 3 * 384;
  float q[3][4], k[3][4], vv[3][4];
  #pragma unroll
  for (int t = 0; t < 3; ++t)
    #pragma unroll
    for (int h = 0; h < 4; ++h) {
      const float* rp = base + t * 384 + h * 32 + lane;
      q[t][h]  = rp[0];
      k[t][h]  = rp[128];
      vv[t][h] = rp[256];
    }
  const float scale = 0.17677669529663687f;  // 1/sqrt(32)
  #pragma unroll
  for (int h = 0; h < 4; ++h) {
    float s[3][3];
    #pragma unroll
    for (int i = 0; i < 3; ++i)
      #pragma unroll
      for (int j = 0; j < 3; ++j)
        s[i][j] = wred32(q[i][h] * k[j][h]) * scale;
    #pragma unroll
    for (int i = 0; i < 3; ++i) {
      float mx = fmaxf(s[i][0], fmaxf(s[i][1], s[i][2]));
      float e0 = expf(s[i][0] - mx), e1 = expf(s[i][1] - mx), e2 = expf(s[i][2] - mx);
      float inv = 1.0f / (e0 + e1 + e2);
      float o = (e0 * vv[0][h] + e1 * vv[1][h] + e2 * vv[2][h]) * inv;
      hout[((size_t)node * 3 + i) * 128 + h * 32 + lane] = o;
    }
  }
}

// ---------------- small elementwise kernels ----------------
__global__ __launch_bounds__(256) void gelu_inplace(float* __restrict__ x, int n) {
  int i = blockIdx.x * 256 + threadIdx.x;
  if (i < n) x[i] = gelu_exact(x[i]);
}

__global__ __launch_bounds__(256) void pack_cg(
    const float* __restrict__ vf, const float* __restrict__ conf,
    float* __restrict__ cg, int rows)
{
  int i = blockIdx.x * 256 + threadIdx.x;
  int total = rows * 129;
  if (i >= total) return;
  int r = i / 129, c = i - r * 129;
  cg[i] = (c < 128) ? vf[(size_t)r * 128 + c] : conf[r];
}

__global__ __launch_bounds__(256) void fuse_gate(
    const float* __restrict__ vf, const float* __restrict__ gpre,
    const float* __restrict__ conf, float* __restrict__ fused, int N)
{
  int i = blockIdx.x * 256 + threadIdx.x;
  if (i >= N * 128) return;
  int n = i >> 7, d = i & 127;
  float acc = 0.f;
  #pragma unroll
  for (int t = 0; t < 3; ++t) {
    size_t r = (size_t)n * 3 + t;
    float gv = 1.0f / (1.0f + expf(-gpre[r * 128 + d]));
    acc += vf[r * 128 + d] * gv * conf[r];
  }
  fused[i] = acc;
}

__global__ __launch_bounds__(256) void deg_count(
    const int* __restrict__ edge, float* __restrict__ deg, int E)
{
  int e = blockIdx.x * 256 + threadIdx.x;
  if (e < E) atomicAdd(&deg[edge[E + e]], 1.0f);
}

__global__ __launch_bounds__(256) void scatter_add(
    const float* __restrict__ x, const int* __restrict__ edge,
    float* __restrict__ agg, int E, int Wf)
{
  long long i = (long long)blockIdx.x * 256 + threadIdx.x;
  if (i >= (long long)E * Wf) return;
  int e = (int)(i / Wf), d = (int)(i % Wf);
  int s = edge[e], t = edge[E + e];
  atomicAdd(&agg[(size_t)t * Wf + d], x[(size_t)s * Wf + d]);
}

__global__ __launch_bounds__(256) void scale_by_deg(
    float* __restrict__ agg, const float* __restrict__ deg, int N, int Wf)
{
  int i = blockIdx.x * 256 + threadIdx.x;
  if (i >= N * Wf) return;
  int n = i / Wf;
  agg[i] /= fmaxf(deg[n], 1.0f);
}

__global__ __launch_bounds__(256) void softmax5(
    const float* __restrict__ logits, float* __restrict__ probs, int N)
{
  int r = blockIdx.x * 256 + threadIdx.x;
  if (r >= N) return;
  float v[5], m = -1e30f;
  #pragma unroll
  for (int c = 0; c < 5; ++c) { v[c] = logits[(size_t)r * 5 + c]; m = fmaxf(m, v[c]); }
  float s = 0.f;
  #pragma unroll
  for (int c = 0; c < 5; ++c) { v[c] = expf(v[c] - m); s += v[c]; }
  float inv = 1.0f / s;
  #pragma unroll
  for (int c = 0; c < 5; ++c) probs[(size_t)r * 5 + c] = v[c] * inv;
}

// ---------------- host side ----------------
extern "C" void kernel_launch(void* const* d_in, const int* in_sizes, int n_in,
                              void* d_out, int out_size, void* d_ws, size_t ws_size,
                              hipStream_t stream) {
  const int N = NNODES, M3 = 3 * N;
  int idx = 0;
  const float* x0   = (const float*)d_in[idx++];
  const float* x1   = (const float*)d_in[idx++];
  const float* x2   = (const float*)d_in[idx++];
  const float* conf = (const float*)d_in[idx++];
  const int*   edge = (const int*)  d_in[idx++];
  const int E = in_sizes[4] / 2;

  // params: JAX pytree flatten order (sorted dict keys, lists in order)
  const float* attn_in_b  = (const float*)d_in[idx++];
  const float* attn_in_w  = (const float*)d_in[idx++];
  const float* attn_out_b = (const float*)d_in[idx++];
  const float* attn_out_w = (const float*)d_in[idx++];
  const float* clf_b = (const float*)d_in[idx++];
  const float* clf_w = (const float*)d_in[idx++];
  const float* emb   = (const float*)d_in[idx++];
  struct Enc { const float *b1, *b2, *be1, *be2, *g1, *g2, *w1, *w2; } enc[3];
  for (int m = 0; m < 3; ++m) {
    enc[m].b1  = (const float*)d_in[idx++];
    enc[m].b2  = (const float*)d_in[idx++];
    enc[m].be1 = (const float*)d_in[idx++];
    enc[m].be2 = (const float*)d_in[idx++];
    enc[m].g1  = (const float*)d_in[idx++];
    enc[m].g2  = (const float*)d_in[idx++];
    enc[m].w1  = (const float*)d_in[idx++];
    enc[m].w2  = (const float*)d_in[idx++];
  }
  const float* ffn_b1 = (const float*)d_in[idx++];
  const float* ffn_b2 = (const float*)d_in[idx++];
  const float* ffn_w1 = (const float*)d_in[idx++];
  const float* ffn_w2 = (const float*)d_in[idx++];
  const float* gate_b = (const float*)d_in[idx++];
  const float* gate_w = (const float*)d_in[idx++];
  struct Gnn { const float *bl, *nb, *ng, *wl, *wr; } gnn[2];
  for (int l = 0; l < 2; ++l) {
    gnn[l].bl = (const float*)d_in[idx++];
    gnn[l].nb = (const float*)d_in[idx++];
    gnn[l].ng = (const float*)d_in[idx++];
    gnn[l].wl = (const float*)d_in[idx++];
    gnn[l].wr = (const float*)d_in[idx++];
  }
  const float* n1_b = (const float*)d_in[idx++];
  const float* n1_g = (const float*)d_in[idx++];
  const float* n2_b = (const float*)d_in[idx++];
  const float* n2_g = (const float*)d_in[idx++];
  const float* res_b = (const float*)d_in[idx++];
  const float* res_w = (const float*)d_in[idx++];

  // workspace layout (floats / dwords)
  float* ws = (float*)d_ws;
  size_t o = 0;
  float* A0    = ws + o; o += (size_t)M3 * 512;  // qkv / ffn-hidden / cg / enc-hidden
  float* VF    = ws + o; o += (size_t)M3 * 128;
  float* T0    = ws + o; o += (size_t)M3 * 128;
  float* FUSED = ws + o; o += (size_t)N * 128;
  float* AGG   = ws + o; o += (size_t)N * 128;
  float* X1    = ws + o; o += (size_t)N * 128;
  float* G1    = ws + o; o += (size_t)N * 128;
  float* X2    = ws + o; o += (size_t)N * 64;
  float* DEG   = ws + o; o += (size_t)N;
  unsigned* ASWZ = (unsigned*)(ws + o); o += (size_t)1875 * 63 * 512 + 512;  // max: x0 swizzled
  unsigned* WSWZ = (unsigned*)(ws + o); o += (size_t)4 * 1024 * 1024;        // weights swizzled

  float* out_logits = (float*)d_out;
  float* out_probs  = out_logits + (size_t)N * 5;
  float* out_emb    = out_probs  + (size_t)N * 5;

  auto swzA = [&](const float* src, int M, int K) {
    int nkt = (K + 31) / 32, Mt = (M + 15) / 16;
    long long total = (long long)Mt * nkt * 256;
    swizzle_pack<<<dim3((unsigned)((total + 255) / 256)), 256, 0, stream>>>(src, ASWZ, M, K, nkt, 0);
  };
  unsigned* wcur = WSWZ;
  auto swzW = [&](const float* w, int Nc, int K) -> unsigned* {
    int nkt = (K + 31) / 32, nt = (Nc + 15) / 16;
    unsigned* dst = wcur;
    wcur += (size_t)nt * nkt * 512;
    long long total = (long long)nt * nkt * 256;
    swizzle_pack<<<dim3((unsigned)((total + 255) / 256)), 256, 0, stream>>>(w, dst, Nc, K, nkt, 1);
    return dst;
  };
  auto gemm = [&](const unsigned* Wsw, const float* bias, const float* add,
                  float* C, int M, int K, int Nc) {
    int nkt = (K + 31) / 32;
    dim3 g((M + 31) / 32, (Nc + 127) / 128);
    gemm_wmma3<<<g, 256, 0, stream>>>((const uint4*)ASWZ, (const uint4*)Wsw, bias, add, C, M, Nc, nkt);
  };
  auto rowop = [&](const float* in, const float* g_, const float* b_,
                   const float* addrow, float* out, int M, int Wd, int stride, int flags) {
    row_op<<<dim3((M + 7) / 8), 256, 0, stream>>>(in, g_, b_, addrow, out, M, Wd, stride, flags);
  };

  // ---- swizzle all weights once per launch ----
  unsigned *w_enc1[3], *w_enc2[3];
  const int dims[3] = {2000, 1000, 800};
  for (int m = 0; m < 3; ++m) {
    w_enc1[m] = swzW(enc[m].w1, 256, dims[m]);
    w_enc2[m] = swzW(enc[m].w2, 128, 256);
  }
  unsigned* w_qkv  = swzW(attn_in_w, 384, 128);
  unsigned* w_aout = swzW(attn_out_w, 128, 128);
  unsigned* w_ff1  = swzW(ffn_w1, 512, 128);
  unsigned* w_ff2  = swzW(ffn_w2, 128, 512);
  unsigned* w_gate = swzW(gate_w, 128, 129);
  unsigned* w_gl0  = swzW(gnn[0].wl, 128, 128);
  unsigned* w_gr0  = swzW(gnn[0].wr, 128, 128);
  unsigned* w_gl1  = swzW(gnn[1].wl, 64, 128);
  unsigned* w_gr1  = swzW(gnn[1].wr, 64, 128);
  unsigned* w_res  = swzW(res_w, 64, 128);
  unsigned* w_clf  = swzW(clf_w, 5, 64);

  // ---- per-modality encoders ----
  const float* xs[3] = {x0, x1, x2};
  for (int m = 0; m < 3; ++m) {
    swzA(xs[m], N, dims[m]);
    gemm(w_enc1[m], enc[m].b1, nullptr, A0, N, dims[m], 256);
    rowop(A0, enc[m].g1, enc[m].be1, nullptr, A0, N, 256, 256, /*LN|GELU*/6);
    swzA(A0, N, 256);
    gemm(w_enc2[m], enc[m].b2, nullptr, T0, N, 256, 128);
    rowop(T0, enc[m].g2, enc[m].be2, emb + m * 128, VF + m * 128, N, 128, 384, /*LN*/2);
  }

  // ---- self-attention over 3 tokens ----
  swzA(VF, M3, 128);
  gemm(w_qkv, attn_in_b, nullptr, A0, M3, 128, 384);                   // qkv [3N,384]
  attn3_kernel<<<dim3((N + 7) / 8), 256, 0, stream>>>(A0, T0, N);      // h [3N,128]
  swzA(T0, M3, 128);
  gemm(w_aout, attn_out_b, VF, A0, M3, 128, 128);                      // vf + proj(h)
  rowop(A0, n1_g, n1_b, nullptr, VF, M3, 128, 128, 2);                 // vf = LN(...)

  // ---- FFN ----
  swzA(VF, M3, 128);
  gemm(w_ff1, ffn_b1, nullptr, A0, M3, 128, 512);
  gelu_inplace<<<dim3((M3 * 512 + 255) / 256), 256, 0, stream>>>(A0, M3 * 512);
  swzA(A0, M3, 512);
  gemm(w_ff2, ffn_b2, VF, T0, M3, 512, 128);                           // vf + ffn
  rowop(T0, n2_g, n2_b, nullptr, VF, M3, 128, 128, 2);                 // vf = LN(...)

  // ---- confidence gate + fuse ----
  pack_cg<<<dim3((M3 * 129 + 255) / 256), 256, 0, stream>>>(VF, conf, A0, M3);
  swzA(A0, M3, 129);
  gemm(w_gate, gate_b, nullptr, T0, M3, 129, 128);
  fuse_gate<<<dim3((N * 128 + 255) / 256), 256, 0, stream>>>(VF, T0, conf, FUSED, N);

  // ---- degrees ----
  (void)hipMemsetAsync(DEG, 0, (size_t)N * sizeof(float), stream);
  deg_count<<<dim3((E + 255) / 256), 256, 0, stream>>>(edge, DEG, E);

  // ---- SAGE layer 1: 128 -> 128 ----
  (void)hipMemsetAsync(AGG, 0, (size_t)N * 128 * sizeof(float), stream);
  scatter_add<<<dim3((int)(((long long)E * 128 + 255) / 256)), 256, 0, stream>>>(FUSED, edge, AGG, E, 128);
  scale_by_deg<<<dim3((N * 128 + 255) / 256), 256, 0, stream>>>(AGG, DEG, N, 128);
  swzA(AGG, N, 128);
  gemm(w_gl0, gnn[0].bl, nullptr, G1, N, 128, 128);
  swzA(FUSED, N, 128);
  gemm(w_gr0, nullptr, G1, G1, N, 128, 128);
  rowop(G1, gnn[0].ng, gnn[0].nb, nullptr, X1, N, 128, 128, /*L2|LN|GELU*/7);

  // ---- SAGE layer 2: 128 -> 64 ----
  (void)hipMemsetAsync(AGG, 0, (size_t)N * 128 * sizeof(float), stream);
  scatter_add<<<dim3((int)(((long long)E * 128 + 255) / 256)), 256, 0, stream>>>(X1, edge, AGG, E, 128);
  scale_by_deg<<<dim3((N * 128 + 255) / 256), 256, 0, stream>>>(AGG, DEG, N, 128);
  swzA(AGG, N, 128);
  gemm(w_gl1, gnn[1].bl, nullptr, G1, N, 128, 64);
  swzA(X1, N, 128);
  gemm(w_gr1, nullptr, G1, G1, N, 128, 64);
  rowop(G1, gnn[1].ng, gnn[1].nb, nullptr, X2, N, 64, 64, 7);

  // ---- residual projection + classifier + softmax ----
  swzA(FUSED, N, 128);
  gemm(w_res, res_b, X2, out_emb, N, 128, 64);
  swzA(out_emb, N, 64);
  gemm(w_clf, clf_b, nullptr, out_logits, N, 64, 5);
  softmax5<<<dim3((N + 255) / 256), 256, 0, stream>>>(out_logits, out_probs, N);
}
